// NetworkNoise18_25477745999971
// MI455X (gfx1250) — matrix operand order repaired and checked by reading, not compiled, not run on hardware
//
#include <hip/hip_runtime.h>
#include <hip/hip_bf16.h>
#include <stdint.h>

typedef __attribute__((ext_vector_type(16))) _Float16 v16h;
typedef __attribute__((ext_vector_type(8)))  _Float16 v8h;
typedef __attribute__((ext_vector_type(8)))  float    v8f;

#define NB 8
#define NT 262144
#define NC 8
#define NK 50
#define NLAYER 18
#define KSTEPS 13           // ceil(400/32)
#define TILE_T 2048
#define MAX_HALO (49*8)
#define FRAG_HALVES (KSTEPS*32*16)   // per-layer weight fragment halves

static __device__ __forceinline__ unsigned lds_off32(const void* p) {
    // addrspace(3)->generic pointers: low 32 bits are the LDS byte offset
    return (unsigned)(uintptr_t)p;
}

// ---------------------------------------------------------------------------
// Setup: weight-norm, fold gating (gw,gb) into extra 8 output columns, and
// write B-matrix fragments in the exact per-lane WMMA register layout.
// B layout used here (16b, 32x16): lane L holds column n=L%16; half j holds k=16*(L/16)+j (A matches).
// Inner index kin = tap*8 + ci ; taps >= 50 are zero padding.
// ---------------------------------------------------------------------------
__global__ void nn18_setup(const float* __restrict__ v0, const float* __restrict__ g0,
                           const float* __restrict__ b0, const float* __restrict__ gw0,
                           const float* __restrict__ gb0, const float* __restrict__ vs,
                           const float* __restrict__ gs, const float* __restrict__ bs,
                           const float* __restrict__ gws, const float* __restrict__ gbs,
                           _Float16* __restrict__ wfrag, float* __restrict__ biasc) {
    __shared__ float norms[NLAYER * NC];
    int tid = threadIdx.x;

    for (int i = tid; i < NLAYER * NC; i += blockDim.x) {
        int l = i / NC, c = i % NC;
        float s = 0.f;
        if (l == 0) {
            for (int k = 0; k < NK; ++k) { float x = v0[c * NK + k]; s += x * x; }
        } else {
            const float* vv = vs + (size_t)((l - 1) * NC + c) * NC * NK;
            for (int j = 0; j < NC * NK; ++j) s += vv[j] * vv[j];
        }
        norms[i] = sqrtf(s);
    }
    __syncthreads();

    auto Wf = [&](int l, int c, int ci, int kt) -> float {
        if (l == 0) {
            if (ci != 0) return 0.f;
            return g0[c] * v0[c * NK + kt] / norms[c];
        }
        int i = l - 1;
        return gs[i * NC + c] *
               vs[(size_t)((i * NC + c) * NC + ci) * NK + kt] / norms[l * NC + c];
    };

    // combined biases: n<8 -> b[n]; n>=8 -> gw.b + gb
    for (int i = tid; i < NLAYER * 16; i += blockDim.x) {
        int l = i / 16, n = i % 16;
        const float* bb = (l == 0) ? b0 : bs + (l - 1) * NC;
        const float* gw = (l == 0) ? gw0 : gws + (l - 1) * NC * NC;
        const float* gb = (l == 0) ? gb0 : gbs + (l - 1) * NC;
        float r;
        if (n < 8) r = bb[n];
        else {
            r = gb[n - 8];
            for (int c = 0; c < NC; ++c) r += gw[(n - 8) * NC + c] * bb[c];
        }
        biasc[l * 16 + n] = r;
    }

    // fragment image: idx = ((l*13+s)*32 + lane)*16 + j
    const int total = NLAYER * FRAG_HALVES;
    for (int idx = tid; idx < total; idx += blockDim.x) {
        int j    = idx & 15;
        int lane = (idx >> 4) & 31;
        int rest = idx >> 9;
        int s    = rest % KSTEPS;
        int l    = rest / KSTEPS;
        int n    = lane & 15;
        int khi  = lane >> 4;
        int kin  = s * 32 + khi * 16 + j;
        int kt = kin >> 3, ci = kin & 7;
        float val = 0.f;
        if (kt < NK) {
            if (n < 8) {
                val = Wf(l, n, ci, kt);
            } else {
                const float* gw = (l == 0) ? gw0 : gws + (l - 1) * NC * NC;
                for (int c = 0; c < NC; ++c) val += gw[(n - 8) * NC + c] * Wf(l, c, ci, kt);
            }
        }
        wfrag[idx] = (_Float16)val;
    }
}

// x [B,1,T] f32 -> h0 [B][T][C] f16, channel 0 = x, others 0
__global__ void nn18_prep(const float* __restrict__ x, _Float16* __restrict__ h0) {
    size_t i = (size_t)blockIdx.x * blockDim.x + threadIdx.x;
    if (i >= (size_t)NB * NT) return;
    v8h z = {};
    z[0] = (_Float16)x[i];
    *(v8h*)(h0 + i * NC) = z;
}

// ---------------------------------------------------------------------------
// One dilated-conv gated-residual layer, WMMA-based.
//   out/gated (N=16 cols) = A(im2col of h, f16) x B(folded weights) + bias
//   h_out = tanh(out)*sigmoid(gated) + h_in
// Dual-accumulator (two 16-t tiles) + sched_group_barrier pipelining.
// ---------------------------------------------------------------------------
__global__ __launch_bounds__(256, 1) void nn18_layer(
        const _Float16* __restrict__ hin, _Float16* __restrict__ hout,
        const _Float16* __restrict__ wfragL, const float* __restrict__ biasL, int d) {
    __shared__ _Float16 stage[(TILE_T + MAX_HALO) * NC];

    const int tb = blockIdx.x % (NT / TILE_T);
    const int b  = blockIdx.x / (NT / TILE_T);
    const long tstart = (long)tb * TILE_T;
    const int halo = 49 * d;
    const int cnt  = TILE_T + halo;          // timesteps to stage
    const _Float16* hinB  = hin  + (size_t)b * NT * NC;
    _Float16*       houtB = hout + (size_t)b * NT * NC;

    // --- stage [tstart-halo, tstart+TILE_T) into LDS (channel-interleaved, 16B/t)
    for (int u = threadIdx.x; u < cnt; u += 256) {
        long t = tstart - halo + u;
        if (t >= 0) {
            unsigned dst = lds_off32(&stage[u * NC]);
            unsigned long long src = (unsigned long long)(uintptr_t)(hinB + t * NC);
            asm volatile("global_load_async_to_lds_b128 %0, %1, off"
                         :: "v"(dst), "v"(src) : "memory");
        } else {
            v8h z = {};
            *(v8h*)(&stage[u * NC]) = z;     // causal zero padding
        }
    }
    asm volatile("s_wait_asynccnt 0x0" ::: "memory");
    __syncthreads();

    const int lane = threadIdx.x & 31;
    const int wave = threadIdx.x >> 5;
    const int m  = lane & 15;
    const int hi = lane >> 4;
    const int co = lane & 7;
    const bool lowHalf = (lane & 8) == 0;

    // per-layer weight fragments -> registers (13 x v16h)
    v16h bw[KSTEPS];
#pragma unroll
    for (int s = 0; s < KSTEPS; ++s)
        bw[s] = ((const v16h*)wfragL)[s * 32 + lane];
    const float biasv = biasL[lane & 15];

    // per-lane A-fragment LDS offsets (in halves), relative to tile base t0*NC
    int offA[KSTEPS], offB[KSTEPS];
#pragma unroll
    for (int s = 0; s < KSTEPS; ++s) {
        int ktA = 4 * s + 2 * hi;      if (ktA > NK - 1) ktA = NK - 1; // zero-wt pad taps
        int ktB = 4 * s + 1 + 2 * hi;  if (ktB > NK - 1) ktB = NK - 1;
        offA[s] = (m - (NK - 1 - ktA) * d) * NC;
        offB[s] = (m - (NK - 1 - ktB) * d) * NC;
    }

    // 128 16-t tiles per block; each wave owns 16, processed two at a time
    for (int ii = 0; ii < TILE_T / 16 / 8; ii += 2) {
        const int tt = wave * (TILE_T / 16 / 8) + ii;
        const int t0 = halo + tt * 16;       // local index of first output t
        const _Float16* tbA = stage + t0 * NC;          // tile tt
        const _Float16* tbB = tbA + 16 * NC;            // tile tt+1

        v8f accA, accB;
#pragma unroll
        for (int j = 0; j < 8; ++j) { accA[j] = biasv; accB[j] = biasv; }

        // 2-deep software pipeline over k-steps, two tiles per step
        v16h bufA[2], bufB[2];
#pragma unroll
        for (int s = 0; s < 2; ++s) {
            v8h a0 = *(const v8h*)(tbA + offA[s]);
            v8h a1 = *(const v8h*)(tbA + offB[s]);
            bufA[s] = __builtin_shufflevector(a0, a1,
                          0,1,2,3,4,5,6,7,8,9,10,11,12,13,14,15);
            v8h b0 = *(const v8h*)(tbB + offA[s]);
            v8h b1 = *(const v8h*)(tbB + offB[s]);
            bufB[s] = __builtin_shufflevector(b0, b1,
                          0,1,2,3,4,5,6,7,8,9,10,11,12,13,14,15);
        }
#pragma unroll
        for (int s = 0; s < KSTEPS; ++s) {
            v16h aA = bufA[s & 1];
            v16h aB = bufB[s & 1];
            if (s + 2 < KSTEPS) {
                v8h a0 = *(const v8h*)(tbA + offA[s + 2]);
                v8h a1 = *(const v8h*)(tbA + offB[s + 2]);
                bufA[s & 1] = __builtin_shufflevector(a0, a1,
                              0,1,2,3,4,5,6,7,8,9,10,11,12,13,14,15);
                v8h b0 = *(const v8h*)(tbB + offA[s + 2]);
                v8h b1 = *(const v8h*)(tbB + offB[s + 2]);
                bufB[s & 1] = __builtin_shufflevector(b0, b1,
                              0,1,2,3,4,5,6,7,8,9,10,11,12,13,14,15);
            }
            accA = __builtin_amdgcn_wmma_f32_16x16x32_f16(
                       false, aA, false, bw[s], (short)0, accA, false, false);
            accB = __builtin_amdgcn_wmma_f32_16x16x32_f16(
                       false, aB, false, bw[s], (short)0, accB, false, false);
        }

        // pin the pipeline: 8-load prologue, then {2 WMMA, 4 DS-read} x11, 4 WMMA tail
        // masks: 0x008 = matrix ops, 0x100 = DS reads
        __builtin_amdgcn_sched_group_barrier(0x100, 8, 0);
#define NN18_STEP  __builtin_amdgcn_sched_group_barrier(0x008, 2, 0); \
                   __builtin_amdgcn_sched_group_barrier(0x100, 4, 0);
        NN18_STEP NN18_STEP NN18_STEP NN18_STEP NN18_STEP NN18_STEP
        NN18_STEP NN18_STEP NN18_STEP NN18_STEP NN18_STEP
#undef NN18_STEP
        __builtin_amdgcn_sched_group_barrier(0x008, 4, 0);

        // D layout: lane n=L%16 (n<8 -> out ch n, n>=8 -> gated ch n-8),
        // vgpr v -> timestep m = v + 8*(L/16). Pair via xor-8 shuffle.
        const long tg = tstart + (long)tt * 16;
#pragma unroll
        for (int v = 0; v < 8; ++v) {
            float mineA = accA[v];
            float partA = __shfl_xor(mineA, 8, 32);
            float outA = lowHalf ? mineA : partA;
            float gatA = lowHalf ? partA : mineA;
            float actA = tanhf(outA) * (1.0f / (1.0f + __expf(-gatA)));
            float mineB = accB[v];
            float partB = __shfl_xor(mineB, 8, 32);
            float outB = lowHalf ? mineB : partB;
            float gatB = lowHalf ? partB : mineB;
            float actB = tanhf(outB) * (1.0f / (1.0f + __expf(-gatB)));
            const int tl = t0 + v + 8 * hi;
            actA += (float)stage[tl * NC + co];               // residual, tile tt
            actB += (float)stage[(tl + 16) * NC + co];        // residual, tile tt+1
            if (lowHalf) {
                houtB[(tg + v + 8 * hi) * NC + co]        = (_Float16)actA;
                houtB[(tg + 16 + v + 8 * hi) * NC + co]   = (_Float16)actB;
            }
        }
    }
}

// means / stds head
__global__ void nn18_head(const _Float16* __restrict__ h,
                          const float* __restrict__ mean_w, const float* __restrict__ mean_b,
                          const float* __restrict__ lv_w,   const float* __restrict__ lv_b,
                          float* __restrict__ out) {
    size_t i = (size_t)blockIdx.x * blockDim.x + threadIdx.x;
    if (i >= (size_t)NB * NT) return;
    v8h hv = *(const v8h*)(h + i * NC);
    float mw = mean_b[0], lv = lv_b[0];
#pragma unroll
    for (int c = 0; c < NC; ++c) {
        float x = (float)hv[c];
        mw += mean_w[c] * x;
        lv += lv_w[c] * x;
    }
    out[i] = mw;
    out[(size_t)NB * NT + i] = __expf(0.5f * lv);
}

extern "C" void kernel_launch(void* const* d_in, const int* in_sizes, int n_in,
                              void* d_out, int out_size, void* d_ws, size_t ws_size,
                              hipStream_t stream) {
    const float* x   = (const float*)d_in[0];
    const float* v0  = (const float*)d_in[1];
    const float* g0  = (const float*)d_in[2];
    const float* b0  = (const float*)d_in[3];
    const float* gw0 = (const float*)d_in[4];
    const float* gb0 = (const float*)d_in[5];
    const float* vs  = (const float*)d_in[6];
    const float* gs  = (const float*)d_in[7];
    const float* bs  = (const float*)d_in[8];
    const float* gws = (const float*)d_in[9];
    const float* gbs = (const float*)d_in[10];
    const float* mean_w = (const float*)d_in[11];
    const float* mean_b = (const float*)d_in[12];
    const float* lv_w   = (const float*)d_in[13];
    const float* lv_b   = (const float*)d_in[14];

    char* ws = (char*)d_ws;
    const size_t HBUF = (size_t)NB * NT * NC * sizeof(_Float16);   // 32 MB
    _Float16* hA    = (_Float16*)ws;
    _Float16* hB    = (_Float16*)(ws + HBUF);
    _Float16* wfrag = (_Float16*)(ws + 2 * HBUF);
    float*    biasc = (float*)(ws + 2 * HBUF + (size_t)NLAYER * FRAG_HALVES * 2);

    nn18_setup<<<1, 256, 0, stream>>>(v0, g0, b0, gw0, gb0, vs, gs, bs, gws, gbs,
                                      wfrag, biasc);
    nn18_prep<<<(NB * NT) / 256, 256, 0, stream>>>(x, hA);

    static const int DIL[NLAYER] = {1,1,1,1,1,1,1,2,2,2,2,4,4,4,4,8,8,8};
    _Float16* bufs[2] = {hA, hB};
    for (int l = 0; l < NLAYER; ++l) {
        nn18_layer<<<NB * (NT / TILE_T), 256, 0, stream>>>(
            bufs[l & 1], bufs[(l + 1) & 1],
            wfrag + (size_t)l * FRAG_HALVES, biasc + l * 16, DIL[l]);
    }
    // after 18 layers (even count) result is back in hA
    nn18_head<<<(NB * NT) / 256, 256, 0, stream>>>(hA, mean_w, mean_b, lv_w, lv_b,
                                                   (float*)d_out);
}
